// RevDecoder_52621939310807
// MI455X (gfx1250) — matrix operand order repaired and checked by reading, not compiled, not run on hardware
//
#include <hip/hip_runtime.h>
#include <hip/hip_bf16.h>

// Problem constants
#define TT 256
#define BB 64
#define HH 1024
#define HHh 512
#define LL 2
#define MAXF 0.875f

#define DI __device__ __forceinline__

typedef __attribute__((ext_vector_type(8)))  __bf16 v8bf;
typedef __attribute__((ext_vector_type(16))) __bf16 v16bf;
typedef __attribute__((ext_vector_type(8)))  float  v8f;

// ---------------------------------------------------------------------------
// Fragment layouts (CDNA5 WMMA 16x16x32 bf16, CDNA5 ISA):
// A (16x32, MxK): lane l -> row m=l&15 ; K = kb..kb+15, kb=(l>>4)*16 (this code's contiguous order, matched by B)
// B (32x16, KxN): pre-packed so lane l reads 16 contiguous bf16 (32 bytes)
// C/D (16x16): lane l, vgpr v -> row = v + 8*(l>>4), col = l&15
// ---------------------------------------------------------------------------

// One contiguous-K segment of the GEMM: single A base/stride -> simple address
// induction (immediate offsets after unroll), B streams linearly from the
// packed weight blocks (1KB per ktile per strip).
template<int MT>
DI void gemm_seg(const __bf16* __restrict__ wblk, int ktCount,
                 const __bf16* __restrict__ A, int lda,
                 int mtbase, int lane, v8f acc[MT]) {
  const int m  = lane & 15;
  const int kb = (lane >> 4) * 16;
  const __bf16* bp  = wblk + lane * 16;
  const __bf16* ap0 = A + (size_t)(mtbase * 16 + m) * lda + kb;
  for (int kt = 0; kt < ktCount; ++kt) {
    __builtin_prefetch(bp + 512, 0, 3);           // near-scope prefetch of next ktile
    union { v8bf h[2]; v16bf v; } ub;
    ub.h[0] = *(const v8bf*)(bp);
    ub.h[1] = *(const v8bf*)(bp + 8);
#pragma unroll
    for (int mt = 0; mt < MT; ++mt) {
      const __bf16* p = ap0 + (size_t)mt * 16 * lda + kt * 32;
      union { v8bf h[2]; v16bf v; } ua;
      ua.h[0] = *(const v8bf*)(p);
      ua.h[1] = *(const v8bf*)(p + 8);
      acc[mt] = __builtin_amdgcn_wmma_f32_16x16x32_bf16(
          false, ua.v, false, ub.v, (short)0, acc[mt], false, false);
    }
    bp += 512;
  }
}

// OUT[64 x N] = concat(A0[64x1024], A1[64x512]) @ W + bias, one 16-wide N-strip,
// MT consecutive 16-row M-tiles starting at mtbase.
template<int MT>
DI void gemm_full(const __bf16* __restrict__ Wp, int strip, int mtbase,
                  const __bf16* __restrict__ A0, int lda0,
                  const __bf16* __restrict__ A1, int lda1,
                  const float* __restrict__ bias, int lane, v8f acc[MT]) {
  const float bv = bias[strip * 16 + (lane & 15)];
#pragma unroll
  for (int mt = 0; mt < MT; ++mt)
#pragma unroll
    for (int v = 0; v < 8; ++v) acc[mt][v] = bv;
  const __bf16* wblk = Wp + (size_t)strip * 48 * 512;
  gemm_seg<MT>(wblk,             32, A0, lda0, mtbase, lane, acc);  // x segment
  gemm_seg<MT>(wblk + 32 * 512,  16, A1, lda1, mtbase, lane, acc);  // h segment
}

DI float sigmoidf_(float x) { return 1.0f / (1.0f + __expf(-x)); }

// Stage A: zr GEMM (N=1024, 64 strips x 2 M-halves = 128 wave-jobs) fused with:
// cols   0..511 -> z = sigmoid*MAXF + (1-MAXF)   (fp32 zbuf)
// cols 512..1023 -> r = sigmoid; rh = bf16(r * hsrc[:, hofs+c])
DI void stage_zr(int gw, const __bf16* Wp, const float* bias,
                 const __bf16* A0, const __bf16* A1,
                 const float* hsrc, int hofs, float* zbuf, __bf16* rh) {
  const int strip = gw >> 1, mtbase = (gw & 1) * 2;
  const int lane  = threadIdx.x & 31;
  v8f acc[2];
  gemm_full<2>(Wp, strip, mtbase, A0, HH, A1, HH, bias, lane, acc);
  const int col16 = lane & 15;
  const int rbase = (lane >> 4) * 8;
#pragma unroll
  for (int mt = 0; mt < 2; ++mt) {
#pragma unroll
    for (int v = 0; v < 8; ++v) {
      int row = (mtbase + mt) * 16 + rbase + v;
      int col = strip * 16 + col16;
      float s = sigmoidf_(acc[mt][v]);
      if (strip < 32) {
        zbuf[row * HHh + col] = s * MAXF + (1.0f - MAXF);
      } else {
        int cc = col - HHh;
        rh[row * HHh + cc] = (__bf16)(s * hsrc[row * HH + hofs + cc]);
      }
    }
  }
}

// Stage B: g GEMM (N=512, 32 strips x 4 M-tiles = 128 wave-jobs) fused with the
// reversible update: hn = z*h_old + (1-z)*tanh(g); fp32 state + bf16 mirror.
DI void stage_g(int gw, const __bf16* Wp, const float* bias,
                const __bf16* A0, const __bf16* A1,
                const float* zbuf, float* state, __bf16* hbfl, int cofs) {
  const int strip = gw >> 2, mtile = gw & 3;
  const int lane  = threadIdx.x & 31;
  v8f acc[1];
  gemm_full<1>(Wp, strip, mtile, A0, HH, A1, HHh, bias, lane, acc);
  const int col16 = lane & 15;
  const int rbase = (lane >> 4) * 8;
#pragma unroll
  for (int v = 0; v < 8; ++v) {
    int row = mtile * 16 + rbase + v;
    int col = strip * 16 + col16;
    float g  = tanhf(acc[0][v]);
    float z  = zbuf[row * HHh + col];
    float ho = state[row * HH + cofs + col];
    float hn = z * ho + (1.0f - z) * g;
    state[row * HH + cofs + col] = hn;
    hbfl[row * HH + cofs + col]  = (__bf16)hn;
  }
}

// Monotonic device-wide barrier (counter zeroed by hipMemsetAsync per launch).
DI void gbar(unsigned* cnt, unsigned& stage) {
  __threadfence();
  __syncthreads();
  ++stage;
  if (threadIdx.x == 0) {
    atomicAdd(cnt, 1u);
    const unsigned target = stage * gridDim.x;
    while (__hip_atomic_load(cnt, __ATOMIC_RELAXED, __HIP_MEMORY_SCOPE_AGENT) < target)
      __builtin_amdgcn_s_sleep(2);
  }
  __syncthreads();
  __threadfence();
}

struct RevParams {
  const float *bzr1, *bg1, *bzr2, *bg2;          // (L,H)/(L,Hh) fp32 biases
  float*   out;                                   // state (L,B,H) fp32 == d_out
  unsigned* sync;
  const __bf16 *xall;                             // (T,B,H) bf16 embedded inputs
  __bf16 *hbf;                                    // (L,B,H) bf16 state mirror
  __bf16 *rh;                                     // (B,Hh)  bf16 r*h scratch
  float*  zbuf;                                   // (B,Hh)  fp32 z scratch
  const __bf16 *Wzr1p, *Wg1p, *Wzr2p, *Wg2p;      // packed bf16 weights, [L][...]
};

__global__ void rev_persist(RevParams p) {
  const int gw = blockIdx.x * (blockDim.x >> 5) + (threadIdx.x >> 5);
  unsigned bar = 0;
  const size_t wzr_l = (size_t)48 * 64 * 512;   // per-layer packed zr weight elems
  const size_t wg_l  = (size_t)48 * 32 * 512;   // per-layer packed g  weight elems
  for (int t = 0; t < TT; ++t) {
    for (int l = 0; l < LL; ++l) {
      const __bf16* x    = (l == 0) ? (p.xall + (size_t)t * BB * HH) : p.hbf;
      __bf16* hbfl       = p.hbf + (size_t)l * BB * HH;
      float*  st         = p.out + (size_t)l * BB * HH;
      const __bf16* Wzr1 = p.Wzr1p + (size_t)l * wzr_l;
      const __bf16* Wg1  = p.Wg1p  + (size_t)l * wg_l;
      const __bf16* Wzr2 = p.Wzr2p + (size_t)l * wzr_l;
      const __bf16* Wg2  = p.Wg2p  + (size_t)l * wg_l;

      // zr1 = [x, h2] @ Wzr1 ; z1 -> zbuf, r1*h2 -> rh
      stage_zr(gw, Wzr1, p.bzr1 + l * HH, x, hbfl + HHh, st, HHh, p.zbuf, p.rh);
      gbar(p.sync, bar);
      // g1 = tanh([x, r1*h2] @ Wg1) ; h1n = z1*h1 + (1-z1)*g1  (cols 0..511)
      stage_g(gw, Wg1, p.bg1 + l * HHh, x, p.rh, p.zbuf, st, hbfl, 0);
      gbar(p.sync, bar);
      // zr2 = [x, h1n] @ Wzr2 ; z2 -> zbuf, r2*h1n -> rh
      stage_zr(gw, Wzr2, p.bzr2 + l * HH, x, hbfl, st, 0, p.zbuf, p.rh);
      gbar(p.sync, bar);
      // g2 = tanh([x, r2*h1n] @ Wg2) ; h2n = z2*h2 + (1-z2)*g2 (cols 512..1023)
      stage_g(gw, Wg2, p.bg2 + l * HHh, x, p.rh, p.zbuf, st, hbfl, HHh);
      gbar(p.sync, bar);
    }
  }
}

// ---------------------------------------------------------------------------
// Setup kernels (fully parallel, run once per launch before the scan)
// ---------------------------------------------------------------------------
__global__ void k_init_state(const float* __restrict__ h0, float* __restrict__ out,
                             __bf16* __restrict__ hbf, int n) {
  int i = blockIdx.x * 256 + threadIdx.x;
  if (i < n) { float v = h0[i]; out[i] = v; hbf[i] = (__bf16)v; }
}

__global__ void k_embed(const int* __restrict__ seq, const float* __restrict__ emb,
                        __bf16* __restrict__ xall, int total) {
  int i = blockIdx.x * 256 + threadIdx.x;
  if (i < total) {
    int tb  = i >> 10;          // t*B + b
    int c   = i & (HH - 1);
    int tok = seq[tb];
    xall[i] = (__bf16)emb[(size_t)tok * HH + c];
  }
}

// Pack W (L,K,N) fp32 row-major -> bf16 blocks in exact B-fragment lane order:
// dst[(l, nt*KT + kt)*512 + lane*16 + j] = W[l][kt*32 + (lane>>4)*16 + j][nt*16 + (lane&15)]
__global__ void k_pack(const float* __restrict__ W, __bf16* __restrict__ dst, int K, int N) {
  const int KT = K >> 5, NT = N >> 4, perL = KT * NT;
  int bid = blockIdx.x;
  int l   = bid / perL;
  int r   = bid % perL;
  int nt  = r / KT;
  int kt  = r % KT;
  const float* Wl = W + (size_t)l * K * N;
  __bf16* dl = dst + ((size_t)l * perL + (size_t)(nt * KT + kt)) * 512;
  for (int e = threadIdx.x; e < 512; e += 256) {
    int lane = e >> 4, j = e & 15;
    int k = kt * 32 + ((lane >> 4) << 4) + j;
    int n = nt * 16 + (lane & 15);
    dl[e] = (__bf16)Wl[(size_t)k * N + n];
  }
}

// ---------------------------------------------------------------------------
extern "C" void kernel_launch(void* const* d_in, const int* in_sizes, int n_in,
                              void* d_out, int out_size, void* d_ws, size_t ws_size,
                              hipStream_t stream) {
  (void)in_sizes; (void)n_in; (void)out_size; (void)ws_size;
  const int*   seq  = (const int*)  d_in[0];
  const float* h0   = (const float*)d_in[1];
  const float* emb  = (const float*)d_in[2];
  const float* Wzr1 = (const float*)d_in[3];
  const float* bzr1 = (const float*)d_in[4];
  const float* Wg1  = (const float*)d_in[5];
  const float* bg1  = (const float*)d_in[6];
  const float* Wzr2 = (const float*)d_in[7];
  const float* bzr2 = (const float*)d_in[8];
  const float* Wg2  = (const float*)d_in[9];
  const float* bg2  = (const float*)d_in[10];
  float* out = (float*)d_out;

  // Workspace layout (256B aligned bump allocator)
  char*  ws  = (char*)d_ws;
  size_t off = 0;
  auto bump = [&](size_t bytes) { char* p = ws + off; off = (off + bytes + 255) & ~(size_t)255; return p; };
  unsigned* sync = (unsigned*)bump(256);
  float*    zbuf = (float*)   bump((size_t)BB * HHh * 4);
  __bf16*   rh   = (__bf16*)  bump((size_t)BB * HHh * 2);
  __bf16*   hbf  = (__bf16*)  bump((size_t)LL * BB * HH * 2);
  __bf16*   xall = (__bf16*)  bump((size_t)TT * BB * HH * 2);
  __bf16*   Wzr1p= (__bf16*)  bump((size_t)LL * 48 * 64 * 512 * 2);
  __bf16*   Wg1p = (__bf16*)  bump((size_t)LL * 48 * 32 * 512 * 2);
  __bf16*   Wzr2p= (__bf16*)  bump((size_t)LL * 48 * 64 * 512 * 2);
  __bf16*   Wg2p = (__bf16*)  bump((size_t)LL * 48 * 32 * 512 * 2);

  hipMemsetAsync(sync, 0, 256, stream);

  { int n = LL * BB * HH;
    k_init_state<<<(n + 255) / 256, 256, 0, stream>>>(h0, out, hbf, n); }
  { int n = TT * BB * HH;
    k_embed<<<(n + 255) / 256, 256, 0, stream>>>(seq, emb, xall, n); }
  k_pack<<<LL * 48 * 64, 256, 0, stream>>>(Wzr1, Wzr1p, HH + HHh, HH);
  k_pack<<<LL * 48 * 32, 256, 0, stream>>>(Wg1,  Wg1p,  HH + HHh, HHh);
  k_pack<<<LL * 48 * 64, 256, 0, stream>>>(Wzr2, Wzr2p, HH + HHh, HH);
  k_pack<<<LL * 48 * 32, 256, 0, stream>>>(Wg2,  Wg2p,  HH + HHh, HHh);

  RevParams p;
  p.bzr1 = bzr1; p.bg1 = bg1; p.bzr2 = bzr2; p.bg2 = bg2;
  p.out = out; p.sync = sync;
  p.xall = xall; p.hbf = hbf; p.rh = rh; p.zbuf = zbuf;
  p.Wzr1p = Wzr1p; p.Wg1p = Wg1p; p.Wzr2p = Wzr2p; p.Wg2p = Wg2p;

  // 16 blocks x 8 waves = 128 waves: every stage (zr: 64 strips x 2 M-halves,
  // g: 32 strips x 4 M-tiles) maps to exactly 128 wave-jobs; persistent across
  // all 2048 dependent stages with a single L2 atomic barrier between stages.
  rev_persist<<<16, 256, 0, stream>>>(p);
}
